// ResNetImpBGDecoder_4363686772811
// MI455X (gfx1250) — compile-verified, matrix operand order checked by reading, not run on hardware
//
#include <hip/hip_runtime.h>
#include <hip/hip_bf16.h>

typedef __attribute__((ext_vector_type(16))) __bf16 v16bf;
typedef __attribute__((ext_vector_type(8)))  __bf16 v8bf;
typedef __attribute__((ext_vector_type(8)))  float  v8f;

#define LSTR 40   // LDS row stride in bf16 elems (32 data + 8 pad; 80B keeps 16B alignment)

__device__ __forceinline__ __bf16 f2bf(float f) {
  unsigned u = __builtin_bit_cast(unsigned, f);
  unsigned r = u + 0x7FFFu + ((u >> 16) & 1u);   // round-to-nearest-even
  unsigned short h = (unsigned short)(r >> 16);
  return __builtin_bit_cast(__bf16, h);
}

// Async copy of one 16B chunk global -> LDS (CDNA5 GLOBAL_LOAD_ASYNC_TO_LDS_B128,
// tracked by ASYNCcnt). VDST VGPR holds the LDS byte address (low 32 bits of the
// generic pointer = wave-relative LDS offset per the aperture mapping).
__device__ __forceinline__ void async_copy16(const void* gptr, void* lptr) {
  unsigned lds_addr = (unsigned)(uintptr_t)lptr;
  asm volatile("global_load_async_to_lds_b128 %0, %1, off"
               :: "v"(lds_addr), "v"(gptr) : "memory");
}
__device__ __forceinline__ void wait_async0() {
  asm volatile("s_wait_asynccnt 0" ::: "memory");
}

// Fragment load from an un-permuted [row][K] LDS tile.
// 16-bit A/B layout: lanes 0-15 hold K {0..7,16..23}, lanes 16-31 hold K {8..15,24..31};
// element e<8 -> K=col0+e, e>=8 -> K=col0+16+(e-8).
__device__ __forceinline__ v16bf load_frag(const __bf16* tile, int rowBase, int lane) {
  int row  = rowBase + (lane & 15);
  int col0 = (lane >> 4) << 3;                  // 0 or 8
  const __bf16* p = tile + row * LSTR + col0;
  v8bf lo = *(const v8bf*)(p);
  v8bf hi = *(const v8bf*)(p + 16);
  return __builtin_shufflevector(lo, hi, 0,1,2,3,4,5,6,7,8,9,10,11,12,13,14,15);
}

// ---------------------------------------------------------------------------
// Implicit-GEMM conv + fused BN / bias / residual / ReLU, bf16 WMMA, f32 acc.
// Block = 128 threads (4 wave32), tile = 64 cout x 64 pixels.
// Weights pre-packed bf16 [tap][cinChunk][coutP][32] (zero padded) and staged
// into LDS with async-to-LDS copies, overlapped with im2col B staging.
// ---------------------------------------------------------------------------
template<int KH,int KW,int STRIDE,int PAD,bool RELU_IN,bool RELU_OUT,bool HAS_RES,bool HAS_BN>
__global__ __launch_bounds__(128) void conv_wmma(
    const float* __restrict__ x, const __bf16* __restrict__ wp,
    const float* __restrict__ bn_g, const float* __restrict__ bn_b,
    const float* __restrict__ bn_m, const float* __restrict__ bn_v,
    const float* __restrict__ res, float* __restrict__ out,
    int B, int Cin, int H, int W, int Cout, int OH, int OW,
    int outCtot, int outCoff, int NCB, int CoutP)
{
  __shared__ alignas(16) __bf16 As[64 * LSTR];
  __shared__ alignas(16) __bf16 Bs[64 * LSTR];
  __shared__ float s_scale[64];
  __shared__ float s_shift[64];

  const int tid  = threadIdx.x;
  const int lane = tid & 31;
  const int wave = tid >> 5;
  const int mBase = wave * 16;            // cout sub-tile of this wave
  const int coutBase = blockIdx.y * 64;
  const int Npix = B * OH * OW;
  const long nBase = (long)blockIdx.x * 64;

  if (tid < 64) {
    int co = coutBase + tid;
    float sc = 1.f, sf = 0.f;
    if (co < Cout) {
      if (HAS_BN) {
        sc = bn_g[co] * rsqrtf(bn_v[co] + 1e-5f);
        sf = bn_b[co] - bn_m[co] * sc;
      } else if (bn_b) {
        sf = bn_b[co];                    // plain bias path (head conv)
      }
    }
    s_scale[tid] = sc; s_shift[tid] = sf;
  }

  v8f acc[4];
#pragma unroll
  for (int i = 0; i < 4; ++i)
#pragma unroll
    for (int j = 0; j < 8; ++j) acc[i][j] = 0.f;

  for (int tap = 0; tap < KH * KW; ++tap) {
    const int kh = tap / KW, kw = tap % KW;
    for (int cbi = 0; cbi < NCB; ++cbi) {
      const int cb = cbi * 32;
      __syncthreads();                    // previous K-step consumption done

      // ---- 1. issue async A-tile copy: 64 rows x 64B = 256 x 16B chunks ----
      {
        const __bf16* abase =
            wp + ((long)(tap * NCB + cbi) * CoutP + coutBase) * 32;
#pragma unroll
        for (int c = 0; c < 2; ++c) {
          int chunk = tid + c * 128;      // 0..255
          int r = chunk >> 2;             // row (cout within tile)
          int p = chunk & 3;              // 16B part within the 64B row
          async_copy16(abase + (long)r * 32 + p * 8, &As[r * LSTR + p * 8]);
        }
      }

      // ---- 2. stage B tile on VALU: im2col + fp32->bf16, [n][k] layout ----
#pragma unroll 4
      for (int e = tid; e < 2048; e += 128) {
        int kk = e >> 6;
        int r  = e & 63;
        long n = nBase + r;
        int cin = cb + kk;
        float f = 0.f;
        if (n < Npix && cin < Cin) {
          int bidx = (int)(n / (OH * OW));
          int rem  = (int)(n % (OH * OW));
          int oh = rem / OW, ow = rem % OW;
          int ih = oh * STRIDE - PAD + kh;
          int iw = ow * STRIDE - PAD + kw;
          if ((unsigned)ih < (unsigned)H && (unsigned)iw < (unsigned)W) {
            f = x[(((long)bidx * Cin + cin) * H + ih) * W + iw];
            if (RELU_IN) f = fmaxf(f, 0.f);
          }
        }
        Bs[r * LSTR + kk] = f2bf(f);
      }

      // ---- 3. close the async + LDS hazards ----
      wait_async0();
      __syncthreads();

      // ---- 4. WMMA ----
      v16bf afrag = load_frag(As, mBase, lane);
#pragma unroll
      for (int nt = 0; nt < 4; ++nt) {
        v16bf bfrag = load_frag(Bs, nt * 16, lane);
        acc[nt] = __builtin_amdgcn_wmma_f32_16x16x32_bf16(
            false, afrag, false, bfrag, (short)0, acc[nt], false, false);
      }
    }
  }

  // ---- epilogue: BN/bias (+residual) (+relu), scatter to NCHW fp32 ----
#pragma unroll
  for (int nt = 0; nt < 4; ++nt) {
    long n = nBase + nt * 16 + (lane & 15);
    if (n >= Npix) continue;
    int bidx = (int)(n / (OH * OW));
    int rem  = (int)(n % (OH * OW));
    int oh = rem / OW, ow = rem % OW;
#pragma unroll
    for (int vv = 0; vv < 8; ++vv) {
      int mt = mBase + ((lane >> 4) << 3) + vv;   // M = vv + 8*(lane/16)
      int co = coutBase + mt;
      if (co >= Cout) continue;
      float val = acc[nt][vv] * s_scale[mt] + s_shift[mt];
      if (HAS_RES) val += res[(((long)bidx * Cout + co) * OH + oh) * OW + ow];
      if (RELU_OUT) val = fmaxf(val, 0.f);
      out[(((long)bidx * outCtot + (outCoff + co)) * OH + oh) * OW + ow] = val;
    }
  }
}

// ---------------------------------------------------------------------------
// Repack fp32 OIHW weights -> bf16 [tap][cinChunk][coutP][32], zero-padded so
// async tile loads never need bounds checks.
// ---------------------------------------------------------------------------
__global__ __launch_bounds__(256) void repack_kernel(
    const float* __restrict__ w, __bf16* __restrict__ wp,
    int Cout, int Cin, int K2, int NCB, int CoutP)
{
  long idx = (long)blockIdx.x * 256 + threadIdx.x;
  long total = (long)K2 * NCB * CoutP * 32;
  if (idx >= total) return;
  int kk = (int)(idx & 31); long t = idx >> 5;
  int co = (int)(t % CoutP); t /= CoutP;
  int cbi = (int)(t % NCB);  int tap = (int)(t / NCB);
  int cin = cbi * 32 + kk;
  __bf16 val = __builtin_bit_cast(__bf16, (unsigned short)0);
  if (co < Cout && cin < Cin)
    val = f2bf(w[((long)co * Cin + cin) * K2 + tap]);
  wp[idx] = val;
}

// ---------------------------------------------------------------------------
// Bilinear upsample (align_corners=True) into a channel-offset of `out`
// ---------------------------------------------------------------------------
__global__ __launch_bounds__(256) void upsample_kernel(
    const float* __restrict__ in, float* __restrict__ out,
    int B, int C, int H, int W, int OH, int OW, int outCtot, int outCoff)
{
  long idx = (long)blockIdx.x * 256 + threadIdx.x;
  long total = (long)B * C * OH * OW;
  if (idx >= total) return;
  int ox = (int)(idx % OW); long t = idx / OW;
  int oy = (int)(t % OH); t /= OH;
  int c = (int)(t % C); int b = (int)(t / C);
  float sy = (OH > 1) ? oy * (float)(H - 1) / (float)(OH - 1) : 0.f;
  float sx = (OW > 1) ? ox * (float)(W - 1) / (float)(OW - 1) : 0.f;
  int y0 = (int)floorf(sy); y0 = y0 < 0 ? 0 : (y0 > H - 1 ? H - 1 : y0);
  int x0 = (int)floorf(sx); x0 = x0 < 0 ? 0 : (x0 > W - 1 ? W - 1 : x0);
  int y1 = y0 + 1 < H ? y0 + 1 : H - 1;
  int x1 = x0 + 1 < W ? x0 + 1 : W - 1;
  float wy = sy - (float)y0, wx = sx - (float)x0;
  const float* p = in + ((long)b * C + c) * H * W;
  float a = p[y0 * W + x0], bb = p[y0 * W + x1];
  float cc = p[y1 * W + x0], dd = p[y1 * W + x1];
  float top = a * (1.f - wx) + bb * wx;
  float bot = cc * (1.f - wx) + dd * wx;
  out[(((long)b * outCtot + outCoff + c) * OH + oy) * OW + ox] = top * (1.f - wy) + bot * wy;
}

// ---------------------------------------------------------------------------
// Head: leave-one-out logsumexp, attention pooling -> [B,21]
// ---------------------------------------------------------------------------
__global__ __launch_bounds__(256) void head_reduce(
    const float* __restrict__ logits, float* __restrict__ out, int HW)
{
  const int NC = 21;
  int b = blockIdx.x;
  int tid = threadIdx.x;
  float accO[NC];
  float accA = 0.f;
#pragma unroll
  for (int k = 0; k < NC; ++k) accO[k] = 0.f;

  const float* base = logits + (long)b * NC * HW;
  for (int p = tid; p < HW; p += 256) {
    float zv[NC];
    float mx = -1e30f;
#pragma unroll
    for (int k = 0; k < NC; ++k) { zv[k] = base[(long)k * HW + p]; mx = fmaxf(mx, zv[k]); }
    float s = 0.f;
#pragma unroll
    for (int k = 0; k < NC; ++k) s += expf(zv[k] - mx);
    float lse = mx + logf(s);

    float l[NC];
    float mxl = -1e30f;
#pragma unroll
    for (int k = 0; k < NC; ++k) {
      float loo = lse + log1pf(-expf(zv[k] - lse));      // may be -inf (exp(-inf)=0 below)
      float a1 = loo, a2 = -zv[k];
      float mm = fmaxf(a1, a2);
      float dd = fminf(a1, a2) - mm;
      float lae = mm + log1pf(expf(dd));                 // logaddexp(loo, -z)
      l[k] = -lae;
      mxl = fmaxf(mxl, l[k]);
    }
    float sl = 0.f;
#pragma unroll
    for (int k = 0; k < NC; ++k) sl += expf(l[k] - mxl);
    float lsel = mxl + logf(sl);
    float attn = 1.f / (1.f + expf(-lsel));
#pragma unroll
    for (int k = 0; k < NC; ++k) accO[k] += attn * l[k];
    accA += attn;
  }

  __shared__ float red[256];
  __shared__ float sO[NC];
  __shared__ float sA;
  red[tid] = accA; __syncthreads();
  for (int s = 128; s > 0; s >>= 1) { if (tid < s) red[tid] += red[tid + s]; __syncthreads(); }
  if (tid == 0) sA = red[0];
  __syncthreads();
  for (int k = 0; k < NC; ++k) {
    red[tid] = accO[k]; __syncthreads();
    for (int s = 128; s > 0; s >>= 1) { if (tid < s) red[tid] += red[tid + s]; __syncthreads(); }
    if (tid == 0) sO[k] = red[0];
    __syncthreads();
  }
  if (tid < NC) out[b * NC + tid] = sO[tid] / sA;
}

// ---------------------------------------------------------------------------
// Host orchestration
// ---------------------------------------------------------------------------
template<int KH,int KW,int S,int P,bool RI,bool RO,bool RES,bool BN>
static inline void launch_conv(hipStream_t st, const float* x, const __bf16* wp,
    const float* g, const float* b, const float* m, const float* v,
    const float* res, float* out,
    int Cin, int H, int W, int Cout, int OH, int OW, int CT, int CO)
{
  int NCB = (Cin + 31) / 32;
  int CoutP = (Cout + 63) & ~63;
  dim3 grid((unsigned)((4 * OH * OW + 63) / 64), (unsigned)((Cout + 63) / 64));
  conv_wmma<KH,KW,S,P,RI,RO,RES,BN><<<grid, 128, 0, st>>>(
      x, wp, g, b, m, v, res, out, 4, Cin, H, W, Cout, OH, OW, CT, CO, NCB, CoutP);
}

#define BNP(i) (const float*)d_in[(i)], (const float*)d_in[(i)+1], \
               (const float*)d_in[(i)+2], (const float*)d_in[(i)+3]

extern "C" void kernel_launch(void* const* d_in, const int* in_sizes, int n_in,
                              void* d_out, int out_size, void* d_ws, size_t ws_size,
                              hipStream_t stream) {
  (void)in_sizes; (void)n_in; (void)out_size; (void)ws_size;
  const float* X = (const float*)d_in[0];

  char* wsb = (char*)d_ws;
  size_t off = 0;
  auto alloc = [&](size_t bytes) -> void* {
    void* p = wsb + off;
    off += (bytes + 255) & ~(size_t)255;
    return p;
  };
  auto repack = [&](int wi, int Cout, int Cin, int K2) -> const __bf16* {
    int NCB = (Cin + 31) / 32;
    int CoutP = (Cout + 63) & ~63;
    long total = (long)K2 * NCB * CoutP * 32;
    __bf16* wp = (__bf16*)alloc((size_t)total * sizeof(__bf16));
    repack_kernel<<<(unsigned)((total + 255) / 256), 256, 0, stream>>>(
        (const float*)d_in[wi], wp, Cout, Cin, K2, NCB, CoutP);
    return wp;
  };

  // ---- weights (bf16, padded async-friendly tiles) ----
  const __bf16* w_stem  = repack(1,   64,   3, 9);
  const __bf16* l1b0w1  = repack(6,   64,  64, 9);
  const __bf16* l1b0w2  = repack(11,  64,  64, 9);
  const __bf16* l1b1w1  = repack(16,  64,  64, 9);
  const __bf16* l1b1w2  = repack(21,  64,  64, 9);
  const __bf16* l2b0w1  = repack(26, 128,  64, 9);
  const __bf16* l2b0w2  = repack(31, 128, 128, 9);
  const __bf16* l2b0ws  = repack(36, 128,  64, 1);
  const __bf16* l2b1w1  = repack(41, 128, 128, 9);
  const __bf16* l2b1w2  = repack(46, 128, 128, 9);
  const __bf16* l3b0w1  = repack(51, 256, 128, 9);
  const __bf16* l3b0w2  = repack(56, 256, 256, 9);
  const __bf16* l3b0ws  = repack(61, 256, 128, 1);
  const __bf16* l3b1w1  = repack(66, 256, 256, 9);
  const __bf16* l3b1w2  = repack(71, 256, 256, 9);
  const __bf16* l4b0w1  = repack(76, 384, 256, 9);
  const __bf16* l4b0w2  = repack(81, 384, 384, 9);
  const __bf16* l4b0ws  = repack(86, 384, 256, 1);
  const __bf16* l4b1w1  = repack(91, 384, 384, 9);
  const __bf16* l4b1w2  = repack(96, 384, 384, 9);
  const __bf16* w_low   = repack(101,128,  64, 1);
  const __bf16* w_dec1  = repack(106,256, 512, 9);
  const __bf16* w_dec2  = repack(111,256, 256, 9);
  const __bf16* w_dec3  = repack(116, 21, 256, 1);

  // ---- activation buffers ----
  const size_t S64 = (size_t)4 * 64 * 128 * 128 * 4;      // 16 MB
  float* A0 = (float*)alloc(S64);
  float* A1 = (float*)alloc(S64);
  float* A2 = (float*)alloc(S64);
  const size_t SL3 = (size_t)4 * 256 * 32 * 32 * 4;       // 4 MB
  float* L3a = (float*)alloc(SL3);
  float* L3b = (float*)alloc(SL3);
  float* L3c = (float*)alloc(SL3);
  const size_t SL4 = (size_t)4 * 384 * 16 * 16 * 4;       // 1.5 MB
  float* L4a = (float*)alloc(SL4);
  float* L4b = (float*)alloc(SL4);
  float* L4c = (float*)alloc(SL4);
  float* CAT = (float*)alloc((size_t)4 * 512 * 128 * 128 * 4); // 128 MB
  float* D1  = (float*)alloc((size_t)4 * 256 * 128 * 128 * 4); // 64 MB
  float* LOGITS = A0;   // A0 is free by decoder time (4*21*128*128*4 < 16MB)

  // ---- stem: conv3x3(3->64) + bn1 (no relu) -> A1 ----
  launch_conv<3,3,1,1,false,false,false,true>(stream, X, w_stem, BNP(2),
      nullptr, A1, 3,128,128, 64,128,128, 64,0);

  // ---- layer1 (64ch, 128x128, identity residuals) ----
  launch_conv<3,3,1,1,false,true ,false,true>(stream, A1, l1b0w1, BNP(7),
      nullptr, A2, 64,128,128, 64,128,128, 64,0);
  launch_conv<3,3,1,1,false,true ,true ,true>(stream, A2, l1b0w2, BNP(12),
      A1, A0, 64,128,128, 64,128,128, 64,0);
  launch_conv<3,3,1,1,false,true ,false,true>(stream, A0, l1b1w1, BNP(17),
      nullptr, A2, 64,128,128, 64,128,128, 64,0);
  launch_conv<3,3,1,1,false,true ,true ,true>(stream, A2, l1b1w2, BNP(22),
      A0, A1, 64,128,128, 64,128,128, 64,0);
  // L1OUT = A1

  // ---- low branch: 1x1(64->128)+bn, written straight into CAT channels [384,512) ----
  launch_conv<1,1,1,0,false,false,false,true>(stream, A1, w_low, BNP(102),
      nullptr, CAT, 64,128,128, 128,128,128, 512,384);

  // ---- layer2 (64->128, stride 2, 64x64) ----
  launch_conv<1,1,2,0,false,false,false,true>(stream, A1, l2b0ws, BNP(37),
      nullptr, A2, 64,128,128, 128,64,64, 128,0);                 // shortcut
  launch_conv<3,3,2,1,false,true ,false,true>(stream, A1, l2b0w1, BNP(27),
      nullptr, A0, 64,128,128, 128,64,64, 128,0);
  launch_conv<3,3,1,1,false,true ,true ,true>(stream, A0, l2b0w2, BNP(32),
      A2, A1, 128,64,64, 128,64,64, 128,0);
  launch_conv<3,3,1,1,false,true ,false,true>(stream, A1, l2b1w1, BNP(42),
      nullptr, A0, 128,64,64, 128,64,64, 128,0);
  launch_conv<3,3,1,1,false,true ,true ,true>(stream, A0, l2b1w2, BNP(47),
      A1, A2, 128,64,64, 128,64,64, 128,0);
  // L2OUT = A2

  // ---- layer3 (128->256, stride 2, 32x32) ----
  launch_conv<1,1,2,0,false,false,false,true>(stream, A2, l3b0ws, BNP(62),
      nullptr, L3a, 128,64,64, 256,32,32, 256,0);
  launch_conv<3,3,2,1,false,true ,false,true>(stream, A2, l3b0w1, BNP(52),
      nullptr, L3b, 128,64,64, 256,32,32, 256,0);
  launch_conv<3,3,1,1,false,true ,true ,true>(stream, L3b, l3b0w2, BNP(57),
      L3a, L3c, 256,32,32, 256,32,32, 256,0);
  launch_conv<3,3,1,1,false,true ,false,true>(stream, L3c, l3b1w1, BNP(67),
      nullptr, L3b, 256,32,32, 256,32,32, 256,0);
  launch_conv<3,3,1,1,false,true ,true ,true>(stream, L3b, l3b1w2, BNP(72),
      L3c, L3a, 256,32,32, 256,32,32, 256,0);
  // L3OUT = L3a

  // ---- layer4 (256->384, stride 2, 16x16) ----
  launch_conv<1,1,2,0,false,false,false,true>(stream, L3a, l4b0ws, BNP(87),
      nullptr, L4a, 256,32,32, 384,16,16, 384,0);
  launch_conv<3,3,2,1,false,true ,false,true>(stream, L3a, l4b0w1, BNP(77),
      nullptr, L4b, 256,32,32, 384,16,16, 384,0);
  launch_conv<3,3,1,1,false,true ,true ,true>(stream, L4b, l4b0w2, BNP(82),
      L4a, L4c, 384,16,16, 384,16,16, 384,0);
  launch_conv<3,3,1,1,false,true ,false,true>(stream, L4c, l4b1w1, BNP(92),
      nullptr, L4b, 384,16,16, 384,16,16, 384,0);
  launch_conv<3,3,1,1,false,true ,true ,true>(stream, L4b, l4b1w2, BNP(97),
      L4c, L4a, 384,16,16, 384,16,16, 384,0);
  // L4OUT = L4a

  // ---- upsample layer4 16x16 -> 128x128 into CAT channels [0,384) ----
  {
    long total = (long)4 * 384 * 128 * 128;
    upsample_kernel<<<(unsigned)((total + 255) / 256), 256, 0, stream>>>(
        L4a, CAT, 4, 384, 16, 16, 128, 128, 512, 0);
  }

  // ---- decoder ----
  launch_conv<3,3,1,1,true ,false,false,true>(stream, CAT, w_dec1, BNP(107),
      nullptr, D1, 512,128,128, 256,128,128, 256,0);              // relu on input
  launch_conv<3,3,1,1,false,false,false,true>(stream, D1, w_dec2, BNP(112),
      nullptr, CAT, 256,128,128, 256,128,128, 256,0);             // reuse CAT
  launch_conv<1,1,1,0,false,false,false,false>(stream, CAT, w_dec3,
      nullptr, (const float*)d_in[117], nullptr, nullptr,         // bias via bn_b slot
      nullptr, LOGITS, 256,128,128, 21,128,128, 21,0);

  // ---- head: LOO-logsumexp attention pooling -> [4,21] ----
  head_reduce<<<4, 256, 0, stream>>>(LOGITS, (float*)d_out, 128 * 128);
}
